// AttentionBlock_43791486550343
// MI455X (gfx1250) — matrix operand order repaired and checked by reading, not compiled, not run on hardware
//
#include <hip/hip_runtime.h>
#include <math.h>

typedef __attribute__((ext_vector_type(16))) _Float16 v16h;
typedef __attribute__((ext_vector_type(8)))  _Float16 v8h;
typedef __attribute__((ext_vector_type(4)))  _Float16 v4h;
typedef __attribute__((ext_vector_type(8)))  float    v8f;

#define ROWS_PER_WAVE   16
#define WAVES_PER_BLOCK 8
#define MAX_BLOCKS      512

// Fragment-store bases (units of one 32x16 f16 B-fragment = 512 halves)
#define FB_WQ   0     //  Wq      N=16  K=64  -> 2
#define FB_WKV  2     //  Wkv     N=16  K=64  -> 2
#define FB_EF1  4     //  eps_fc1 N=128 K=128 -> 32
#define FB_EF2  36    //  eps_fc2 N=32  K=128 -> 8
#define FB_EF   44    //  eps_fc  N=16  K=32  -> 1
#define FB_RF1  45    //  res_fc1 N=128 K=80(->96) -> 24
#define FB_RM   69    //  res_match                -> 24
#define FB_RF2  93    //  res_fc2 N=32  K=128 -> 8
#define FB_RF   101   //  res_fc  N=64  K=32  -> 4
#define NFRAGS  105

struct Consts {
  float cq[8], ck[8], cv[8], bq[8], bk[8], bv[8];
  float Ah[2], Ch[2];   // pre-scaled by 0.5*log2(e): softmax slope = Ah*q + Ch
  float G[16];          // [e*2 + h]
  float bias2[8];
  float ef1b[128], eln1s[128], eln1b[128];
  float ef2b[32],  eln2s[32],  eln2b[32];
  float efb[16];
  float rf1b[128], rmb[128], rln1s[128], rln1b[128];
  float rf2b[32],  rln2s[32], rln2b[32];
  float rfb[64];
};

struct WaveScratch {
  _Float16 sQ[3 * 512];    // A-frag layout: frags 0,1 = query (K 0..63), frag 2 = eps_|pad
  _Float16 sX[4 * 512];    // A-frag layout generic staging (up to K=128)
  float    acc1[16 * 128];
  float    acc2[16 * 128];
  float    q[16 * 16];
  float    kv[16 * 16];
  float    s[32];          // attention S values [h*16 + t]
};

struct Smem {
  _Float16    frags[NFRAGS * 512];
  Consts      c;
  WaveScratch w[WAVES_PER_BLOCK];
};  // ~318 KB (<= 320KB WGP LDS)

struct Params {
  const float *query, *kv, *Wq, *Wkv, *wq_embed, *wkv_embed;
  const float *in_proj_w, *in_proj_b, *out_proj_w, *out_proj_b;
  const float *ef1w, *ef1b, *eln1s, *eln1b;
  const float *ef2w, *ef2b, *eln2s, *eln2b;
  const float *efw,  *efb;
  const float *rf1w, *rf1b, *rln1s, *rln1b;
  const float *rmw,  *rmb;
  const float *rf2w, *rf2b, *rln2s, *rln2b;
  const float *rfw,  *rfb;
  float* out;
  int tilesTotal;
};

// Branchless tanh-form GELU: exp2 + rcp transcendentals only.
// gelu(x) ~= 0.5*x*(1 + tanh(sqrt(2/pi)*(x + 0.044715 x^3)))
// tanh(u) = 1 - 2/(exp(2u)+1);  2u*log2(e) folded into the polynomial.
__device__ inline float gelu_fast(float x) {
  const float t  = x * x;
  const float c  = fmaf(t, 0.10294166f, 2.30211830f);  // 2*log2e*sqrt(2/pi)*(1,0.044715)
  const float e  = exp2f(x * c);
  const float rc = __builtin_amdgcn_rcpf(e + 1.0f);
  const float th = fmaf(-2.0f, rc, 1.0f);
  return 0.5f * x * (1.0f + th);
}

// Position of element (m, k) inside an A-fragment-layout buffer.
__device__ inline int afrag_off(int m, int k) {
  const int kc   = k >> 5;
  const int kl   = k & 31;
  const int half = (kl >> 3) & 1;
  const int i    = ((kl >> 1) & 8) | (kl & 7);
  return kc * 512 + (m + (half << 4)) * 16 + i;
}

__device__ inline v16h load_afrag(const _Float16* a, int kc, int lane) {
  return *(const v16h*)(a + kc * 512 + lane * 16);
}
__device__ inline v16h load_bfrag(const _Float16* frags, int f, int lane) {
  return *(const v16h*)(frags + f * 512 + lane * 16);
}

// GEMM with the layer bias folded into the WMMA C operand (bias depends only on
// the output column n, which is fixed per lane per tile -> broadcast into all 8 slots).
template <int KC, int NT>
__device__ inline void gemm(const _Float16* afr, const _Float16* frags, int fbase,
                            float* acc, int ldn, const float* bias, int lane) {
  v16h af[KC];
#pragma unroll
  for (int kc = 0; kc < KC; ++kc) af[kc] = load_afrag(afr, kc, lane);
#pragma unroll
  for (int nt = 0; nt < NT; ++nt) {
    const int n  = nt * 16 + (lane & 15);
    const int rb = (lane >> 4) * 8;
    const float bv = bias ? bias[n] : 0.0f;
    v8f d = {bv, bv, bv, bv, bv, bv, bv, bv};
#pragma unroll
    for (int kc = 0; kc < KC; ++kc) {
      v16h b = load_bfrag(frags, fbase + nt * KC + kc, lane);
      d = __builtin_amdgcn_wmma_f32_16x16x32_f16(false, af[kc], false, b, (short)0, d,
                                                 false, false);
    }
#pragma unroll
    for (int r = 0; r < 8; ++r) acc[(rb + r) * ldn + n] = d[r];
  }
}

// Two GEMMs sharing the same A fragments (res_fc1 + res_match), biases folded.
template <int KC, int NT>
__device__ inline void gemm_dual(const _Float16* afr, const _Float16* frags,
                                 int b1, float* o1, const float* bias1,
                                 int b2, float* o2, const float* bias2,
                                 int ldn, int lane) {
  v16h af[KC];
#pragma unroll
  for (int kc = 0; kc < KC; ++kc) af[kc] = load_afrag(afr, kc, lane);
#pragma unroll
  for (int nt = 0; nt < NT; ++nt) {
    const int n  = nt * 16 + (lane & 15);
    const int rb = (lane >> 4) * 8;
    const float bv1 = bias1[n], bv2 = bias2[n];
    v8f d1 = {bv1, bv1, bv1, bv1, bv1, bv1, bv1, bv1};
    v8f d2 = {bv2, bv2, bv2, bv2, bv2, bv2, bv2, bv2};
#pragma unroll
    for (int kc = 0; kc < KC; ++kc) {
      v16h w1 = load_bfrag(frags, b1 + nt * KC + kc, lane);
      d1 = __builtin_amdgcn_wmma_f32_16x16x32_f16(false, af[kc], false, w1, (short)0, d1,
                                                  false, false);
      v16h w2 = load_bfrag(frags, b2 + nt * KC + kc, lane);
      d2 = __builtin_amdgcn_wmma_f32_16x16x32_f16(false, af[kc], false, w2, (short)0, d2,
                                                  false, false);
    }
#pragma unroll
    for (int r = 0; r < 8; ++r) {
      o1[(rb + r) * ldn + n] = d1[r];
      o2[(rb + r) * ldn + n] = d2[r];
    }
  }
}

// LayerNorm + fast GELU (+ optional residual), all 32 lanes busy:
// lane = (half, m); each half handles N/2 features, combined via shfl_xor(16).
// acc already includes the layer bias (folded into WMMA C).
__device__ inline void ln_gelu_frag(const float* acc, int N,
                                    const float* lns, const float* lnb,
                                    const float* radd, _Float16* dstfrag, int lane) {
  const int m  = lane & 15;
  const int jb = (lane >> 4) * (N >> 1);
  const int je = jb + (N >> 1);
  float s1 = 0.f, s2 = 0.f;
  for (int j = jb; j < je; j += 4) {
    const float4 x4 = *(const float4*)(acc + m * N + j);
    s1 += x4.x + x4.y + x4.z + x4.w;
    s2 += x4.x * x4.x + x4.y * x4.y + x4.z * x4.z + x4.w * x4.w;
  }
  s1 += __shfl_xor(s1, 16, 32);
  s2 += __shfl_xor(s2, 16, 32);
  const float mu  = s1 / N;                     // N is a power of two: exact mul
  const float var = s2 / N - mu * mu;
  const float inv = rsqrtf(fmaxf(var, 0.f) + 1e-5f);
  for (int j = jb; j < je; j += 8) {
    v8h o;
#pragma unroll
    for (int u = 0; u < 8; ++u) {
      const float x = acc[m * N + j + u];
      float g = gelu_fast((x - mu) * inv * lns[j + u] + lnb[j + u]);
      if (radd) g += radd[m * N + j + u];
      o[u] = (_Float16)g;
    }
    *(v8h*)(dstfrag + afrag_off(m, j)) = o;
  }
}

__device__ inline v8h pack8(const float4 a, const float4 b) {
  v8h o;
  o[0] = (_Float16)a.x; o[1] = (_Float16)a.y; o[2] = (_Float16)a.z; o[3] = (_Float16)a.w;
  o[4] = (_Float16)b.x; o[5] = (_Float16)b.y; o[6] = (_Float16)b.z; o[7] = (_Float16)b.w;
  return o;
}

extern __shared__ char smem_raw[];

__global__ __launch_bounds__(256) void fused_attn_mlp(Params p) {
  Smem* sm = (Smem*)smem_raw;
  const int tid = threadIdx.x;
  const int nth = blockDim.x;

  // ---------- per-block init: weights -> f16 B-fragments (8-elem vector chunks) ----
  {
    struct WT { const float* w; int K, base, chunks; };
    const WT tbl[9] = {
        {p.Wq, 64, FB_WQ, 2 * 64},      {p.Wkv, 64, FB_WKV, 2 * 64},
        {p.ef1w, 128, FB_EF1, 32 * 64}, {p.ef2w, 128, FB_EF2, 8 * 64},
        {p.efw, 32, FB_EF, 1 * 64},     {p.rf1w, 80, FB_RF1, 24 * 64},
        {p.rmw, 80, FB_RM, 24 * 64},    {p.rf2w, 128, FB_RF2, 8 * 64},
        {p.rfw, 32, FB_RF, 4 * 64}};
    for (int t = 0; t < 9; ++t) {
      const WT wt   = tbl[t];
      const int kch = (wt.K + 31) / 32;
      for (int c = tid; c < wt.chunks; c += nth) {
        const int f = c >> 6, rem = c & 63, ln = rem >> 1, i0 = (rem & 1) * 8;
        const int nt2 = f / kch, kc = f % kch;
        const int n  = nt2 * 16 + (ln & 15);
        const int k0 = kc * 32 + 2 * i0 + 8 * (ln >> 4);
        v8h v;
        if (k0 < wt.K) {
          const float4 f0 = *(const float4*)(wt.w + n * wt.K + k0);
          const float4 f1 = *(const float4*)(wt.w + n * wt.K + k0 + 4);
          v = pack8(f0, f1);
        } else {
          v = (v8h){};
        }
        *(v8h*)(sm->frags + (wt.base + f) * 512 + ln * 16 + i0) = v;
      }
    }
  }
  Consts& C = sm->c;
  for (int i = tid; i < 128; i += nth) {
    C.ef1b[i] = p.ef1b[i]; C.eln1s[i] = p.eln1s[i]; C.eln1b[i] = p.eln1b[i];
    C.rf1b[i] = p.rf1b[i]; C.rmb[i] = p.rmb[i];
    C.rln1s[i] = p.rln1s[i]; C.rln1b[i] = p.rln1b[i];
  }
  for (int i = tid; i < 32; i += nth) {
    C.ef2b[i] = p.ef2b[i]; C.eln2s[i] = p.eln2s[i]; C.eln2b[i] = p.eln2b[i];
    C.rf2b[i] = p.rf2b[i]; C.rln2s[i] = p.rln2s[i]; C.rln2b[i] = p.rln2b[i];
  }
  for (int i = tid; i < 64; i += nth) C.rfb[i] = p.rfb[i];
  for (int i = tid; i < 16; i += nth) C.efb[i] = p.efb[i];
  if (tid == 0) {
    float cq[8], ck[8], cv[8], bq[8], bk[8], bv[8];
    for (int e = 0; e < 8; ++e) {
      float a = 0.f, b = 0.f, c = 0.f;
      for (int e2 = 0; e2 < 8; ++e2) {
        a += p.in_proj_w[e * 8 + e2]        * p.wq_embed[e2];
        b += p.in_proj_w[(8 + e) * 8 + e2]  * p.wkv_embed[e2];
        c += p.in_proj_w[(16 + e) * 8 + e2] * p.wkv_embed[e2];
      }
      cq[e] = a; ck[e] = b; cv[e] = c;
      bq[e] = p.in_proj_b[e];
      bk[e] = p.in_proj_b[8 + e];
      bv[e] = p.in_proj_b[16 + e];
    }
    const float HL2E = 0.72134752044448170f;  // 0.5 * log2(e)
    for (int h = 0; h < 2; ++h) {
      float A = 0.f, Cc = 0.f;
      for (int d = 0; d < 4; ++d) {
        const int ix = h * 4 + d;
        A  += cq[ix] * ck[ix];
        Cc += bq[ix] * ck[ix];
      }
      C.Ah[h] = A * HL2E;   // pre-scaled log2-domain softmax slope coefficients
      C.Ch[h] = Cc * HL2E;
    }
    for (int e = 0; e < 8; ++e) {
      for (int h = 0; h < 2; ++h) {
        float g = 0.f;
        for (int d = 0; d < 4; ++d) g += p.out_proj_w[e * 8 + h * 4 + d] * cv[h * 4 + d];
        C.G[e * 2 + h] = g;
      }
      float b2 = p.out_proj_b[e];
      for (int e2 = 0; e2 < 8; ++e2) b2 += p.out_proj_w[e * 8 + e2] * bv[e2];
      C.bias2[e] = b2;
    }
  }
  __syncthreads();

  // ---------- per-wave pipeline: grid-stride over 16-row tiles ----------
  const int wave = tid >> 5;
  const int lane = tid & 31;
  WaveScratch* W = &sm->w[wave];
  const int slot  = blockIdx.x * WAVES_PER_BLOCK + wave;
  const int slots = gridDim.x * WAVES_PER_BLOCK;

  for (int tile = slot; tile < p.tilesTotal; tile += slots) {
    const long rowBase = (long)tile * ROWS_PER_WAVE;

    // Stage query (-> sQ frags 0,1) and kv (-> sX frags 0,1), f16 A-frag layout.
    for (int c = lane; c < 128; c += 32) {       // 128 chunks of 8 elements
      const int m = c >> 3, k0 = (c & 7) * 8;
      const float* qp = p.query + (rowBase + m) * 64 + k0;
      const float* kp = p.kv    + (rowBase + m) * 64 + k0;
      *(v8h*)(W->sQ + afrag_off(m, k0)) =
          pack8(*(const float4*)qp, *(const float4*)(qp + 4));
      *(v8h*)(W->sX + afrag_off(m, k0)) =
          pack8(*(const float4*)kp, *(const float4*)(kp + 4));
    }

    // q = Q @ Wq^T ; kv = KV @ Wkv^T  (no bias in the reference)
    gemm<2, 1>(W->sQ, sm->frags, FB_WQ, W->q, 16, nullptr, lane);
    gemm<2, 1>(W->sX, sm->frags, FB_WKV, W->kv, 16, nullptr, lane);

    // Attention, fully collapsed. Logits are AFFINE in kv_j: the constant part
    // cancels in softmax, so only the log2-domain slope a2 = Ah*q + Ch remains.
    {
      const int   h  = lane >> 4, t = lane & 15;
      const float Ah = C.Ah[h], Ch = C.Ch[h];
      const int   idx0 = lane * 4;               // feature chunk this lane writes
      const int   t2 = idx0 >> 3;
      const int   e0 = idx0 & 7;
      float Ga[4], Gb[4], Bb[4];                 // hoisted output coefficients
#pragma unroll
      for (int u = 0; u < 4; ++u) {
        Ga[u] = C.G[(e0 + u) * 2];
        Gb[u] = C.G[(e0 + u) * 2 + 1];
        Bb[u] = C.bias2[e0 + u];
      }
#pragma unroll 1
      for (int r = 0; r < 16; ++r) {
        const float4 k0 = *(const float4*)(W->kv + r * 16);
        const float4 k1 = *(const float4*)(W->kv + r * 16 + 4);
        const float4 k2 = *(const float4*)(W->kv + r * 16 + 8);
        const float4 k3 = *(const float4*)(W->kv + r * 16 + 12);
        const float kvv[16] = {k0.x, k0.y, k0.z, k0.w, k1.x, k1.y, k1.z, k1.w,
                               k2.x, k2.y, k2.z, k2.w, k3.x, k3.y, k3.z, k3.w};
        const float qv = W->q[r * 16 + t];
        const float a2 = fmaf(Ah, qv, Ch);       // log2-domain slope
        float tj[16], mx = -3.0e38f;
#pragma unroll
        for (int j = 0; j < 16; ++j) {
          tj[j] = a2 * kvv[j];
          mx = fmaxf(mx, tj[j]);
        }
        float se = 0.f, sk = 0.f;
#pragma unroll
        for (int j = 0; j < 16; ++j) {
          const float ev = exp2f(tj[j] - mx);
          se += ev;
          sk = fmaf(ev, kvv[j], sk);
        }
        W->s[lane] = sk * __builtin_amdgcn_rcpf(se);  // S_{h,t}
        const float S0 = W->s[t2], S1 = W->s[16 + t2];
        float4 vf; v4h vh;
        vf.x = fmaf(Ga[0], S0, fmaf(Gb[0], S1, Bb[0]));
        vf.y = fmaf(Ga[1], S0, fmaf(Gb[1], S1, Bb[1]));
        vf.z = fmaf(Ga[2], S0, fmaf(Gb[2], S1, Bb[2]));
        vf.w = fmaf(Ga[3], S0, fmaf(Gb[3], S1, Bb[3]));
        vh[0] = (_Float16)vf.x; vh[1] = (_Float16)vf.y;
        vh[2] = (_Float16)vf.z; vh[3] = (_Float16)vf.w;
        *(float4*)(W->acc2 + r * 128 + idx0) = vf;   // f32 eps for residual
        *(v4h*)(W->sX + afrag_off(r, idx0))  = vh;   // f16 A for eps_fc1
      }
    }

    // eps MLP (biases folded into WMMA C)
    gemm<4, 8>(W->sX, sm->frags, FB_EF1, W->acc1, 128, C.ef1b, lane);
    ln_gelu_frag(W->acc1, 128, C.eln1s, C.eln1b, W->acc2, W->sX, lane);
    gemm<4, 2>(W->sX, sm->frags, FB_EF2, W->acc1, 32, C.ef2b, lane);
    ln_gelu_frag(W->acc1, 32, C.eln2s, C.eln2b, nullptr, W->sX, lane);

    // eps_ = h @ eps_fc_w^T + b -> sQ frag 2 (K 64..79); zero-pad K 80..95.
    {
      const v16h a0 = load_afrag(W->sX, 0, lane);
      const v16h b0 = load_bfrag(sm->frags, FB_EF, lane);
      const int n = lane & 15, rb = (lane >> 4) * 8;
      const float bv = C.efb[n];
      v8f d = {bv, bv, bv, bv, bv, bv, bv, bv};
      d = __builtin_amdgcn_wmma_f32_16x16x32_f16(false, a0, false, b0, (short)0, d,
                                                 false, false);
      {
        const int m = lane >> 1, kz = 80 + (lane & 1) * 8;
        *(v8h*)(W->sQ + afrag_off(m, kz)) = (v8h){};
      }
#pragma unroll
      for (int r = 0; r < 8; ++r)
        W->sQ[afrag_off(rb + r, 64 + n)] = (_Float16)d[r];
    }

    // res trunk: inputs = sQ frags (query|eps_|pad); fc1 and match share A frags.
    gemm_dual<3, 8>(W->sQ, sm->frags, FB_RF1, W->acc1, C.rf1b, FB_RM, W->acc2, C.rmb,
                    128, lane);
    ln_gelu_frag(W->acc1, 128, C.rln1s, C.rln1b, W->acc2, W->sX, lane);
    gemm<4, 2>(W->sX, sm->frags, FB_RF2, W->acc1, 32, C.rf2b, lane);
    ln_gelu_frag(W->acc1, 32, C.rln2s, C.rln2b, nullptr, W->sX, lane);

    // out = h @ res_fc_w^T + b (bias folded): stage in acc1, coalesced float4 stores.
    gemm<1, 4>(W->sX, sm->frags, FB_RF, W->acc1, 64, C.rfb, lane);
    for (int e = lane; e < 256; e += 32) {
      const int r = e >> 4, c4 = (e & 15) * 4;
      const float4 v = *(const float4*)(W->acc1 + r * 64 + c4);
      *(float4*)(p.out + (rowBase + r) * 64 + c4) = v;
    }
  }
}

extern "C" void kernel_launch(void* const* d_in, const int* in_sizes, int n_in,
                              void* d_out, int out_size, void* d_ws, size_t ws_size,
                              hipStream_t stream) {
  (void)n_in; (void)out_size; (void)d_ws; (void)ws_size;
  Params p;
  p.query      = (const float*)d_in[0];
  p.kv         = (const float*)d_in[1];
  p.Wq         = (const float*)d_in[2];
  p.Wkv        = (const float*)d_in[3];
  p.wq_embed   = (const float*)d_in[4];
  p.wkv_embed  = (const float*)d_in[5];
  p.in_proj_w  = (const float*)d_in[6];
  p.in_proj_b  = (const float*)d_in[7];
  p.out_proj_w = (const float*)d_in[8];
  p.out_proj_b = (const float*)d_in[9];
  p.ef1w  = (const float*)d_in[10];
  p.ef1b  = (const float*)d_in[11];
  p.eln1s = (const float*)d_in[12];
  p.eln1b = (const float*)d_in[13];
  p.ef2w  = (const float*)d_in[14];
  p.ef2b  = (const float*)d_in[15];
  p.eln2s = (const float*)d_in[16];
  p.eln2b = (const float*)d_in[17];
  p.efw   = (const float*)d_in[18];
  p.efb   = (const float*)d_in[19];
  p.rf1w  = (const float*)d_in[20];
  p.rf1b  = (const float*)d_in[21];
  p.rln1s = (const float*)d_in[22];
  p.rln1b = (const float*)d_in[23];
  p.rmw   = (const float*)d_in[24];
  p.rmb   = (const float*)d_in[25];
  p.rf2w  = (const float*)d_in[26];
  p.rf2b  = (const float*)d_in[27];
  p.rln2s = (const float*)d_in[28];
  p.rln2b = (const float*)d_in[29];
  p.rfw   = (const float*)d_in[30];
  p.rfb   = (const float*)d_in[31];
  p.out   = (float*)d_out;

  const int nrows = in_sizes[0] / 64;                 // B = MC*BATCH
  p.tilesTotal    = nrows / ROWS_PER_WAVE;            // tiles of 16 rows
  int blocks = (p.tilesTotal + WAVES_PER_BLOCK - 1) / WAVES_PER_BLOCK;
  if (blocks > MAX_BLOCKS) blocks = MAX_BLOCKS;       // grid-stride amortizes init
  const size_t sh = sizeof(Smem);                     // ~311 KB dynamic LDS (<=320KB/WGP)

  (void)hipFuncSetAttribute((const void*)fused_attn_mlp,
                            hipFuncAttributeMaxDynamicSharedMemorySize, (int)sh);
  fused_attn_mlp<<<blocks, 256, sh, stream>>>(p);
}
